// CrossModalAttention_44650480009699
// MI455X (gfx1250) — compile-verified, matrix operand order checked by reading, not run on hardware
//
#include <hip/hip_runtime.h>
#include <hip/hip_bf16.h>

typedef _Float16 half_t;
typedef __attribute__((ext_vector_type(16))) _Float16 v16h;
typedef __attribute__((ext_vector_type(8)))  float    v8f;
typedef __attribute__((ext_vector_type(4))) unsigned int v4u;
typedef __attribute__((ext_vector_type(8))) int          v8i_;
typedef __attribute__((ext_vector_type(4))) int          v4i_;

#define B_SZ  8
#define N_SZ  2048
#define D_SZ  512
#define H_SZ  8
#define DH_SZ 64
#define SCALE 0.125f   // 64^-0.5

// ---------------------------------------------------------------------------
// WMMA fragment load: 16x32 f16, A-layout (B-layout identical when the source
// is stored [n][k] contiguous-in-k).
// lane l: row = l&15, hi = l>>4; elems 0..7 -> k = hi*8+i ; 8..15 -> 16+hi*8+i
// ---------------------------------------------------------------------------
__device__ __forceinline__ v16h load_frag(const half_t* __restrict__ base, int ld) {
  const int lane = threadIdx.x & 31;
  const int row  = lane & 15;
  const int hi   = lane >> 4;
  const half_t* p = base + (size_t)row * ld + hi * 8;
  v16h f;
#pragma unroll
  for (int i = 0; i < 8; ++i) f[i] = p[i];
#pragma unroll
  for (int i = 0; i < 8; ++i) f[8 + i] = p[16 + i];
  return f;
}

__device__ __forceinline__ v8f wmma_f16(v16h a, v16h b, v8f c) {
  return __builtin_amdgcn_wmma_f32_16x16x32_f16(false, a, false, b, (short)0, c, false, false);
}

// ---------------------------------------------------------------------------
// Tensor Data Mover: DMA a 2-D tile of 2-byte elements from global to LDS.
// D# packed per CDNA5 ISA 8.3/8.4: group0 = {count=1, lds_addr, global_addr,
// type=2}; group1 = {data_size=2B, tensor_dim0/1, tile_dim0/1, dim0_stride}.
// 1-D transfers: tile1 = 0 (unused).  All descriptor dwords must be SGPRs ->
// force with readfirstlane (values are wave-uniform by construction).
// ---------------------------------------------------------------------------
__device__ __forceinline__ unsigned lds_off(const void* p) {
  return (unsigned)(size_t)p;   // low 32 bits of shared-aperture addr = LDS byte offset
}

__device__ __forceinline__ void tdm_load(unsigned ldsOff, const void* g,
                                         unsigned tile0, unsigned tile1,
                                         unsigned tdim0, unsigned tdim1,
                                         unsigned stride0) {
  size_t ga = (size_t)g;
  unsigned lo = (unsigned)__builtin_amdgcn_readfirstlane((int)(unsigned)ga);
  unsigned hi = (unsigned)__builtin_amdgcn_readfirstlane((int)(unsigned)(ga >> 32));
  ldsOff      = (unsigned)__builtin_amdgcn_readfirstlane((int)ldsOff);
  v4u g0 = { 1u,                                   // count=1, user mode
             ldsOff,                               // lds_addr (bytes)
             lo,                                   // global_addr[31:0]
             (hi & 0x01FFFFFFu) | (2u << 30) };    // global_addr[56:32] | type=2
  v8i_ g1;
  g1[0] = (int)(1u << 16);                                   // data_size = 2 bytes
  g1[1] = (int)((tdim0 & 0xFFFFu) << 16);                    // tensor_dim0[15:0] @bit48
  g1[2] = (int)((tdim0 >> 16) | ((tdim1 & 0xFFFFu) << 16));  // tdim0 hi | tensor_dim1 lo
  g1[3] = (int)((tdim1 >> 16) | ((tile0 & 0xFFFFu) << 16));  // tdim1 hi | tile_dim0
  g1[4] = (int)(tile1 & 0xFFFFu);                            // tile_dim1 (tile_dim2=0)
  g1[5] = (int)stride0;                                      // tensor_dim0_stride[31:0]
  g1[6] = 0;                                                 // stride0 hi | stride1 lo
  g1[7] = 0;
  v4i_ z4 = {0, 0, 0, 0};
#if __clang_major__ >= 23
  v8i_ z8 = {0, 0, 0, 0, 0, 0, 0, 0};
  __builtin_amdgcn_tensor_load_to_lds(g0, g1, z4, z4, z8, 0);
#else
  __builtin_amdgcn_tensor_load_to_lds(g0, g1, z4, z4, 0);
#endif
}

// ---------------------------------------------------------------------------
// Prep kernels
// ---------------------------------------------------------------------------
__global__ void cvt_f32_to_f16(const float* __restrict__ src, half_t* __restrict__ dst, int nelem) {
  int i = blockIdx.x * blockDim.x + threadIdx.x;
  if (i < nelem) dst[i] = (half_t)src[i];
}

// WT[n][k] = (f16) W[k][n], W is 512x512 row-major f32
__global__ void transpose_w(const float* __restrict__ W, half_t* __restrict__ WT) {
  __shared__ float tile[16][17];
  const int k0 = blockIdx.x * 16;
  const int n0 = blockIdx.y * 16;
  const int tx = threadIdx.x, ty = threadIdx.y;
  tile[ty][tx] = W[(size_t)(k0 + ty) * D_SZ + n0 + tx];
  __syncthreads();
  WT[(size_t)(n0 + ty) * D_SZ + k0 + tx] = (half_t)tile[tx][ty];
}

// ---------------------------------------------------------------------------
// QKV projection: block = 8 waves, computes a 32x512 slab of x@W.
// A-tile (32x512 f16, contiguous 32KB) DMA'd to LDS once by TDM and shared by
// all waves; each wave owns 64 columns -> 2 A-frags x 4 B-frags -> 8 WMMAs
// per 32-wide k-step.  Q gets SCALE folded in.  V stored transposed.
// ---------------------------------------------------------------------------
__global__ void __launch_bounds__(256) qkv_kernel(
    const half_t* __restrict__ xh,
    const half_t* __restrict__ WqT, const half_t* __restrict__ WkT,
    const half_t* __restrict__ WvT,
    half_t* __restrict__ Q, half_t* __restrict__ K, half_t* __restrict__ Vt) {
  __shared__ half_t Alds[32 * D_SZ];
  const int wave  = threadIdx.x >> 5;
  const int mtile = blockIdx.x;              // 0..511  (B*N/32)
  const int which = blockIdx.y;              // 0=Q 1=K 2=V
  const half_t* WT = (which == 0) ? WqT : (which == 1) ? WkT : WvT;
  const int lane = threadIdx.x & 31;
  const int n = lane & 15, hi = lane >> 4;

  if (wave == 0) {
    tdm_load(lds_off(Alds), xh + (size_t)mtile * 32 * D_SZ,
             /*tile*/ 32 * D_SZ, 0, /*tensor*/ 32 * D_SZ, 1, 32 * D_SZ);
    __builtin_amdgcn_s_wait_tensorcnt((short)0);
  }
  __syncthreads();

  v8f acc[2][4] = {};
  const half_t* brow = WT + (size_t)wave * 64 * D_SZ;
#pragma unroll 2
  for (int k0 = 0; k0 < D_SZ; k0 += 32) {
    v16h a0 = load_frag(Alds + k0, D_SZ);
    v16h a1 = load_frag(Alds + 16 * D_SZ + k0, D_SZ);
#pragma unroll
    for (int j = 0; j < 4; ++j) {
      v16h b = load_frag(brow + (size_t)j * 16 * D_SZ + k0, D_SZ);
      acc[0][j] = wmma_f16(a0, b, acc[0][j]);
      acc[1][j] = wmma_f16(a1, b, acc[1][j]);
    }
  }

#pragma unroll
  for (int ms = 0; ms < 2; ++ms) {
#pragma unroll
    for (int j = 0; j < 4; ++j) {
#pragma unroll
      for (int r = 0; r < 8; ++r) {
        const int m   = mtile * 32 + ms * 16 + 8 * hi + r;
        const int col = wave * 64 + j * 16 + n;
        const int b   = m >> 11, seq = m & (N_SZ - 1);
        const int h   = col >> 6, dh = col & 63;
        const int bh  = b * H_SZ + h;
        if (which == 0)
          Q[((size_t)bh * N_SZ + seq) * DH_SZ + dh] = (half_t)(acc[ms][j][r] * SCALE);
        else if (which == 1)
          K[((size_t)bh * N_SZ + seq) * DH_SZ + dh] = (half_t)acc[ms][j][r];
        else
          Vt[((size_t)bh * DH_SZ + dh) * N_SZ + seq] = (half_t)acc[ms][j][r];
      }
    }
  }
}

// ---------------------------------------------------------------------------
// Flash attention: one wave per 16-query tile per (b,h); 32 keys per step.
// K tile (32x64, contiguous) and V^T tile (64 rows x 32 keys, stride 2048)
// are double-buffered into LDS with TDM; next tile's DMA overlaps current
// tile's 8 WMMAs + online softmax (s_wait_tensorcnt 2 keeps 1 tile in flight).
// ---------------------------------------------------------------------------
__global__ void __launch_bounds__(128) attn_kernel(
    const half_t* __restrict__ Q, const half_t* __restrict__ K,
    const half_t* __restrict__ Vt, const float* __restrict__ mask,
    half_t* __restrict__ attnout) {
  __shared__ half_t Kl[4][2][32 * DH_SZ];   // 32 keys x 64 dh
  __shared__ half_t Vl[4][2][DH_SZ * 32];   // 64 dh rows x 32 keys
  __shared__ half_t Pl[4][16 * 32];
  const int wave = threadIdx.x >> 5;
  const int lane = threadIdx.x & 31;
  const int n = lane & 15, hi = lane >> 4;
  const int qtile = blockIdx.x * 4 + wave;   // 0..8191
  const int bh = qtile >> 7;
  const int qt = qtile & 127;
  const int b  = bh >> 3;
  const int h  = bh & 7;
  const int qbase = qt * 16;

  const half_t* qptr = Q + ((size_t)bh * N_SZ + qbase) * DH_SZ;
  v16h qa0 = load_frag(qptr, DH_SZ);
  v16h qa1 = load_frag(qptr + 32, DH_SZ);

  v8f o0 = {}, o1 = {}, o2 = {}, o3 = {};
  float mrow[8], lsum[8];
#pragma unroll
  for (int r = 0; r < 8; ++r) { mrow[r] = -1e30f; lsum[r] = 0.f; }

  const float* mbase = mask + ((size_t)b * N_SZ + qbase) * N_SZ;
  half_t* pl = &Pl[wave][0];

  auto issue = [&](int buf, int kb) {
    tdm_load(lds_off(&Kl[wave][buf][0]),
             K + ((size_t)bh * N_SZ + kb) * DH_SZ,
             /*tile*/ 32 * DH_SZ, 0, /*tensor*/ 32 * DH_SZ, 1, 32 * DH_SZ);
    tdm_load(lds_off(&Vl[wave][buf][0]),
             Vt + (size_t)bh * DH_SZ * N_SZ + kb,
             /*tile*/ 32, DH_SZ, /*tensor*/ 32, DH_SZ, /*stride*/ N_SZ);
  };

  issue(0, 0);
  for (int kb = 0; kb < N_SZ; kb += 32) {
    const int cur = (kb >> 5) & 1;
    if (kb + 32 < N_SZ) {
      issue(cur ^ 1, kb + 32);
      __builtin_amdgcn_s_wait_tensorcnt((short)2);   // current tile's 2 DMAs done
    } else {
      __builtin_amdgcn_s_wait_tensorcnt((short)0);
    }

    // S(16x32) = Q @ K^T
    const half_t* kl = &Kl[wave][cur][0];
    v8f s0 = {}, s1 = {};
    {
      v16h kf0 = load_frag(kl, DH_SZ);
      v16h kf1 = load_frag(kl + 32, DH_SZ);
      s0 = wmma_f16(qa0, kf0, s0);
      s0 = wmma_f16(qa1, kf1, s0);
      v16h kf2 = load_frag(kl + 16 * DH_SZ, DH_SZ);
      v16h kf3 = load_frag(kl + 16 * DH_SZ + 32, DH_SZ);
      s1 = wmma_f16(qa0, kf2, s1);
      s1 = wmma_f16(qa1, kf3, s1);
    }

    // multiplicative mask + online softmax
    float p[16];
#pragma unroll
    for (int r = 0; r < 8; ++r) {
      const float* mr = mbase + (size_t)(8 * hi + r) * N_SZ + kb;
      float v0 = s0[r] * (1.f + mr[n]);
      float v1 = s1[r] * (1.f + mr[16 + n]);
      float rm = fmaxf(v0, v1);
#pragma unroll
      for (int off = 1; off < 16; off <<= 1)
        rm = fmaxf(rm, __shfl_xor(rm, off, 32));
      float nm    = fmaxf(mrow[r], rm);
      float alpha = __expf(mrow[r] - nm);
      mrow[r] = nm;
      float p0 = __expf(v0 - nm);
      float p1 = __expf(v1 - nm);
      p[r] = p0; p[8 + r] = p1;
      float rs = p0 + p1;
#pragma unroll
      for (int off = 1; off < 16; off <<= 1)
        rs += __shfl_xor(rs, off, 32);
      lsum[r] = lsum[r] * alpha + rs;
      o0[r] *= alpha; o1[r] *= alpha; o2[r] *= alpha; o3[r] *= alpha;
    }

    // C-layout -> A-layout via per-wave LDS slice
#pragma unroll
    for (int r = 0; r < 8; ++r) {
      pl[(8 * hi + r) * 32 + n]      = (half_t)p[r];
      pl[(8 * hi + r) * 32 + 16 + n] = (half_t)p[8 + r];
    }
    v16h pf = load_frag(pl, 32);

    // O(16x64) += P(16x32) @ V(32x64); Vl rows are dh, contiguous in key
    const half_t* vl = &Vl[wave][cur][0];
    v16h vf0 = load_frag(vl, 32);
    o0 = wmma_f16(pf, vf0, o0);
    v16h vf1 = load_frag(vl + 16 * 32, 32);
    o1 = wmma_f16(pf, vf1, o1);
    v16h vf2 = load_frag(vl + 32 * 32, 32);
    o2 = wmma_f16(pf, vf2, o2);
    v16h vf3 = load_frag(vl + 48 * 32, 32);
    o3 = wmma_f16(pf, vf3, o3);
  }

#pragma unroll
  for (int r = 0; r < 8; ++r) {
    float inv = 1.f / lsum[r];
    size_t row = (size_t)b * N_SZ + qbase + 8 * hi + r;
    half_t* op = attnout + row * (H_SZ * DH_SZ) + h * DH_SZ + n;
    op[0]  = (half_t)(o0[r] * inv);
    op[16] = (half_t)(o1[r] * inv);
    op[32] = (half_t)(o2[r] * inv);
    op[48] = (half_t)(o3[r] * inv);
  }
}

// ---------------------------------------------------------------------------
// Output projection: same slab structure as qkv (TDM A-tile, 32x64 per wave),
// f32 output with bias.
// ---------------------------------------------------------------------------
__global__ void __launch_bounds__(256) out_proj_kernel(
    const half_t* __restrict__ A, const half_t* __restrict__ WoT,
    const float* __restrict__ bo, float* __restrict__ out) {
  __shared__ half_t Alds[32 * D_SZ];
  const int wave  = threadIdx.x >> 5;
  const int mtile = blockIdx.x;
  const int lane = threadIdx.x & 31;
  const int n = lane & 15, hi = lane >> 4;

  if (wave == 0) {
    tdm_load(lds_off(Alds), A + (size_t)mtile * 32 * D_SZ,
             32 * D_SZ, 0, 32 * D_SZ, 1, 32 * D_SZ);
    __builtin_amdgcn_s_wait_tensorcnt((short)0);
  }
  __syncthreads();

  v8f acc[2][4] = {};
  const half_t* brow = WoT + (size_t)wave * 64 * D_SZ;
#pragma unroll 2
  for (int k0 = 0; k0 < D_SZ; k0 += 32) {
    v16h a0 = load_frag(Alds + k0, D_SZ);
    v16h a1 = load_frag(Alds + 16 * D_SZ + k0, D_SZ);
#pragma unroll
    for (int j = 0; j < 4; ++j) {
      v16h b = load_frag(brow + (size_t)j * 16 * D_SZ + k0, D_SZ);
      acc[0][j] = wmma_f16(a0, b, acc[0][j]);
      acc[1][j] = wmma_f16(a1, b, acc[1][j]);
    }
  }

#pragma unroll
  for (int ms = 0; ms < 2; ++ms) {
#pragma unroll
    for (int j = 0; j < 4; ++j) {
      const int col = wave * 64 + j * 16 + n;
      const float bias = bo[col];
#pragma unroll
      for (int r = 0; r < 8; ++r) {
        size_t m = (size_t)mtile * 32 + ms * 16 + 8 * hi + r;
        out[m * D_SZ + col] = acc[ms][j][r] + bias;
      }
    }
  }
}

// ---------------------------------------------------------------------------
extern "C" void kernel_launch(void* const* d_in, const int* in_sizes, int n_in,
                              void* d_out, int out_size, void* d_ws, size_t ws_size,
                              hipStream_t stream) {
  const float* x    = (const float*)d_in[0];
  const float* mask = (const float*)d_in[1];
  const float* Wq   = (const float*)d_in[2];
  const float* Wk   = (const float*)d_in[3];
  const float* Wv   = (const float*)d_in[4];
  const float* Wo   = (const float*)d_in[5];
  const float* bo   = (const float*)d_in[6];
  float* out = (float*)d_out;

  const size_t XE = (size_t)B_SZ * N_SZ * D_SZ;  // 8,388,608 elems
  const size_t WE = (size_t)D_SZ * D_SZ;         // 262,144 elems
  half_t* ws   = (half_t*)d_ws;
  half_t* xh   = ws;  ws += XE;
  half_t* WqT  = ws;  ws += WE;
  half_t* WkT  = ws;  ws += WE;
  half_t* WvT  = ws;  ws += WE;
  half_t* WoT  = ws;  ws += WE;
  half_t* Qb   = ws;  ws += XE;
  half_t* Kb   = ws;  ws += XE;
  half_t* Vtb  = ws;  ws += XE;
  half_t* attn = ws;  ws += XE;

  cvt_f32_to_f16<<<(int)((XE + 255) / 256), 256, 0, stream>>>(x, xh, (int)XE);
  dim3 tb(16, 16);
  transpose_w<<<dim3(32, 32), tb, 0, stream>>>(Wq, WqT);
  transpose_w<<<dim3(32, 32), tb, 0, stream>>>(Wk, WkT);
  transpose_w<<<dim3(32, 32), tb, 0, stream>>>(Wv, WvT);
  transpose_w<<<dim3(32, 32), tb, 0, stream>>>(Wo, WoT);

  qkv_kernel<<<dim3(512, 3), 256, 0, stream>>>(xh, WqT, WkT, WvT, Qb, Kb, Vtb);
  attn_kernel<<<dim3(2048), 128, 0, stream>>>(Qb, Kb, Vtb, mask, attn);
  out_proj_kernel<<<dim3(512), 256, 0, stream>>>(attn, WoT, bo, out);
}
